// HarmonicOscillator_61211873902607
// MI455X (gfx1250) — matrix operand order repaired and checked by reading, not compiled, not run on hardware
//
#include <hip/hip_runtime.h>
#include <hip/hip_bf16.h>
#include <math.h>

#define LF 200                 // frames
#define FS 960                 // FRAME_SIZE
#define LW (LF * FS)           // 192000 samples
#define NB 4                   // batch
#define NH 32                  // harmonics
#define NC 192                 // channels
#define INV_SR (1.0f / 48000.0f)
#define TWO_PI    6.28318530717958647692f
#define INV_2PI   0.15915494309189533577f
#define PI_F      3.14159265358979323846f
#define HALF_PI_F 1.57079632679489661923f

typedef __attribute__((ext_vector_type(16))) _Float16 v16h;
typedef __attribute__((ext_vector_type(8)))  float    v8f;

// ---------------------------------------------------------------------------
// Kernel 1: amps[n,h,l] = exp( sum_c W[h,c] * x[n,c,l] + bias[h] )
// GEMM M=32 x N=800 x K=192 on the matrix pipe (v_wmma_f32_16x16x32_f16).
// One wave per (mtile, ntile); 2 x 50 tiles; 6 k-steps of 32.
// ---------------------------------------------------------------------------
__global__ __launch_bounds__(32)
void k_amps_wmma(const float* __restrict__ W, const float* __restrict__ x,
                 const float* __restrict__ bias, float* __restrict__ amps)
{
    const int lane = threadIdx.x & 31;
    const int r    = lane & 15;     // row/col within tile
    const int hs   = lane >> 4;     // half-wave select
    const int ntile = blockIdx.x;   // 0..49 -> 16 columns each
    const int mtile = blockIdx.y;   // 0..1  -> 16 harmonics each

    const int colg = ntile * 16 + r;          // 0..799 (n*LF + l)
    const int n    = colg / LF;
    const int l    = colg - n * LF;
    const int mg   = mtile * 16 + r;          // harmonic row for A loads

    v8f acc = {};
    for (int ks = 0; ks < NC; ks += 32) {
        // A fragment 16x32 f16 (ISA 7.12.2): lanes 0-15 hold K=0..7,16..23;
        // lanes 16-31 hold K=8..15,24..31 (pairs per VGPR).
        v16h a;
        #pragma unroll
        for (int e = 0; e < 16; ++e) {
            int k = (e < 8) ? (8 * hs + e) : (16 + 8 * hs + (e - 8));
            a[e] = (_Float16)W[mg * NC + ks + k];
        }
        // B fragment 32x16 f16: lanes 0-15 hold K=0..7,16..23; lanes 16-31 K=8..15,24..31.
        v16h bf;
        #pragma unroll
        for (int e = 0; e < 16; ++e) {
            int k = (e < 8) ? (8 * hs + e) : (16 + 8 * hs + (e - 8));
            bf[e] = (_Float16)x[(n * NC + (ks + k)) * LF + l];
        }
        acc = __builtin_amdgcn_wmma_f32_16x16x32_f16(
            /*neg_a=*/false, a, /*neg_b=*/false, bf,
            /*c_mod=*/(short)0, acc, /*reuse_a=*/false, /*reuse_b=*/false);
    }
    // D layout: VGPR v holds M = v + 8*hs, N = r.
    #pragma unroll
    for (int v = 0; v < 8; ++v) {
        int h = mtile * 16 + v + 8 * hs;
        amps[(n * NH + h) * LF + l] = expf(acc[v] + bias[h]);
    }
}

// ---------------------------------------------------------------------------
// Kernel 2: base_phase[n,i] = inclusive cumsum_i( interp(f0)[n,i] / SR ).
// Two-pass per-block scan: 750-sample segment sums -> shared scan -> rewalk.
// interp(f0*mul) == mul*interp(f0), so a single cumsum per batch suffices.
// ---------------------------------------------------------------------------
__device__ __forceinline__ float f0_interp(const float* __restrict__ f, int i) {
    float c = (i + 0.5f) * (1.0f / FS) - 0.5f;
    c = fminf(fmaxf(c, 0.0f), (float)(LF - 1));
    int lo = (int)c;                     // c >= 0 -> trunc == floor
    int hi = min(lo + 1, LF - 1);
    float w = c - (float)lo;
    return f[lo] * (1.0f - w) + f[hi] * w;
}

__global__ __launch_bounds__(256)
void k_phase_scan(const float* __restrict__ f0, float* __restrict__ base)
{
    __shared__ float sh[256];
    const int n = blockIdx.x;
    const int t = threadIdx.x;
    const int SEG = LW / 256;            // 750
    const float* f = f0 + n * LF;

    // pass 1: segment sums
    float s = 0.0f;
    const int i0 = t * SEG;
    for (int k = 0; k < SEG; ++k)
        s += f0_interp(f, i0 + k) * INV_SR;
    sh[t] = s;
    __syncthreads();

    // inclusive Hillis-Steele scan over 256 partials
    for (int off = 1; off < 256; off <<= 1) {
        float v = (t >= off) ? sh[t - off] : 0.0f;
        __syncthreads();
        sh[t] += v;
        __syncthreads();
    }
    float run = (t == 0) ? 0.0f : sh[t - 1];   // exclusive offset

    // pass 2: write inclusive prefix
    float* o = base + n * LW;
    for (int k = 0; k < SEG; ++k) {
        int i = i0 + k;
        run += f0_interp(f, i) * INV_SR;
        o[i] = run;
    }
}

// ---------------------------------------------------------------------------
// Kernel 3 (bandwidth-bound): per sample i, for all 32 harmonics:
//   theta = (h+1) * 2pi * (base[i]-base[0])
//   phi   = triangle-fold(theta) in [-pi/2, pi/2]  (== asin(sin(theta)))
//   sin(theta) == sin(phi); wave = mean_h(sin * interp(amps))
// Coalesced plane-strided phi stores; amps loads are frame-granular (cached).
// ---------------------------------------------------------------------------
__global__ __launch_bounds__(256)
void k_synth(const float* __restrict__ base, const float* __restrict__ amps,
             float* __restrict__ wave, float* __restrict__ phi)
{
    const int idx = blockIdx.x * 256 + threadIdx.x;     // over NB*LW = 768000
    if (idx >= NB * LW) return;
    const int n = idx / LW;
    const int i = idx - n * LW;

    const float tb = TWO_PI * (base[n * LW + i] - base[n * LW]);

    float c = (i + 0.5f) * (1.0f / FS) - 0.5f;
    c = fminf(fmaxf(c, 0.0f), (float)(LF - 1));
    const int   lo = (int)c;
    const int   hi = min(lo + 1, LF - 1);
    const float w  = c - (float)lo;

    const float* an = amps + n * NH * LF;
    float* pn = phi + (size_t)n * NH * LW + i;

    float accw = 0.0f;
    #pragma unroll 4
    for (int h = 0; h < NH; ++h) {
        float amp   = an[h * LF + lo] * (1.0f - w) + an[h * LF + hi] * w;
        float theta = tb * (float)(h + 1);
        float r     = theta - TWO_PI * rintf(theta * INV_2PI);   // [-pi, pi]
        float p     = (fabsf(r) > HALF_PI_F) ? (copysignf(PI_F, r) - r) : r;
        float s     = __sinf(p);                                 // |p| <= pi/2
        accw += s * amp;
        pn[(size_t)h * LW] = p;
    }
    wave[n * LW + i] = accw * (1.0f / NH);
}

// ---------------------------------------------------------------------------
extern "C" void kernel_launch(void* const* d_in, const int* in_sizes, int n_in,
                              void* d_out, int out_size, void* d_ws, size_t ws_size,
                              hipStream_t stream) {
    // setup_inputs order: x [4,192,200], f0 [4,1,200], W [32,192], b [32]
    const float* x    = (const float*)d_in[0];
    const float* f0   = (const float*)d_in[1];
    const float* W    = (const float*)d_in[2];
    const float* bias = (const float*)d_in[3];

    // outputs concatenated: wave [4,1,192000] then phi [4,32,192000]
    float* wave = (float*)d_out;
    float* phi  = wave + NB * LW;

    // workspace: amps [4,32,200] then base_phase [4,192000]
    float* amps = (float*)d_ws;
    float* base = amps + NB * NH * LF;

    k_amps_wmma<<<dim3(50, 2), 32, 0, stream>>>(W, x, bias, amps);
    k_phase_scan<<<NB, 256, 0, stream>>>(f0, base);
    k_synth<<<(NB * LW + 255) / 256, 256, 0, stream>>>(base, amps, wave, phi);
    (void)in_sizes; (void)n_in; (void)out_size; (void)ws_size;
}
